// STAttentionBlock_23029614641814
// MI455X (gfx1250) — matrix operand order repaired and checked by reading, not compiled, not run on hardware
//
#include <hip/hip_runtime.h>

// ---------------------------------------------------------------------------
// ST-Attention block for MI455X (gfx1250, wave32, WMMA + TDM).
// All channel contractions run through v_wmma_f32_16x16x32_bf16 with f32
// accumulate; activations/weights staged in bf16 to halve HBM traffic.
// A-tiles are staged LDS-side by the Tensor Data Mover (tensor_load_to_lds,
// TENSORcnt); B-tiles are transposed into LDS by the waves.
// ---------------------------------------------------------------------------

typedef __attribute__((ext_vector_type(16))) __bf16 v16bf;
typedef __attribute__((ext_vector_type(8)))  __bf16 v8bf;
typedef __attribute__((ext_vector_type(8)))  float  v8f;
typedef __attribute__((ext_vector_type(4)))  unsigned int u32x4;
typedef __attribute__((ext_vector_type(8)))  int i32x8;
typedef __attribute__((ext_vector_type(4)))  int i32x4;

#define TPX   10800          // T*V pixels per (n, channel) row
#define TDIM  400
#define VDIM  27
#define LDSS  40             // padded LDS row stride (bf16 elems), BK=32 + 8 pad

union FragU { v16bf v; v8bf h[2]; };

// A-fragment (16x32 bf16, M x K): lane m<16 holds row m, K 0..7 & 16..23;
// lane m+16 holds row m, K 8..15 & 24..31  (ISA 7.12.2 16-bit A layout).
__device__ inline v16bf load_fragA(const __bf16* s, int rowBase, int lane) {
  const int row = rowBase + (lane & 15);
  const int kh  = (lane >> 4) << 3;              // 0 or 8
  FragU u;
  const __bf16* p = s + row * LDSS + kh;
  u.h[0] = *(const v8bf*)(p);
  u.h[1] = *(const v8bf*)(p + 16);
  return u.v;
}

// B-fragment (32x16 bf16, K x N) loaded from LDS tile stored as B^T (N-major):
// lane n<16 holds column n, K 0..7 & 16..23; lane n+16 holds column n, K 8..15 & 24..31.
__device__ inline v16bf load_fragB(const __bf16* s, int rowBase, int lane) {
  const int row = rowBase + (lane & 15);
  const int kh  = (lane >> 4) << 3;              // 0 or 8
  FragU u;
  const __bf16* p = s + row * LDSS + kh;
  u.h[0] = *(const v8bf*)(p);
  u.h[1] = *(const v8bf*)(p + 16);
  return u.v;
}

// ---------------------------------------------------------------------------
// TDM: DMA a (128 rows x 32 bf16) tile of a row-major (256 x kdim) matrix
// into LDS, hardware-padding each 64B row to 80B (matches LDSS=40).
// D# built per CDNA5 ISA ch.8: group0 {count=1, lds_addr, global_addr, type=2},
// group1 {data_size=2B, pad 16DW->+4DW, dims/strides}. 2D tensor -> groups
// 2/3 zero. This toolchain's builtin takes 6 args (extra int32x8 group).
// ---------------------------------------------------------------------------
__device__ inline void tdm_load_A_tile(unsigned int lds_byte, unsigned long long gaddr,
                                       int kdim) {
  u32x4 g0;
  g0.x = 1u;                                        // count=1, user mode
  g0.y = lds_byte;                                  // LDS dest byte address
  g0.z = (unsigned int)gaddr;                       // global_addr[31:0]
  g0.w = (unsigned int)(gaddr >> 32) | (2u << 30);  // global_addr[56:32] | type=2

  i32x8 g1;
  g1[0] = (1 << 16)        // data_size = 1 -> 2 bytes
        | (1 << 20)        // pad_enable
        | (3 << 22)        // pad_interval code 3 = 16 DWORDs (64B) of data
        | (3 << 25);       // pad_amount  code 3 = 4 DWORDs (16B) of pad
  g1[1] = (kdim & 0xFFFF) << 16;        // tensor_dim0[15:0]  @ bits 63:48
  g1[2] = (kdim >> 16) | (256 << 16);   // tensor_dim0[31:16] | tensor_dim1[15:0]
  g1[3] = (32 << 16);                   // tensor_dim1[31:16]=0 | tile_dim0=32
  g1[4] = 128;                          // tile_dim1=128, tile_dim2=0
  g1[5] = kdim;                         // tensor_dim0_stride[31:0]
  g1[6] = 0;
  g1[7] = 0;

  i32x4 z4 = {0, 0, 0, 0};
  i32x8 z8 = {0, 0, 0, 0, 0, 0, 0, 0};
  __builtin_amdgcn_tensor_load_to_lds(g0, g1, z4, z4, z8, 0);
}

// ---------------------------------------------------------------------------
// Generic bf16 GEMM:  D[o, j] = sum_k A[o, k] * B_n[k, j]   per batch n.
//   MODE 0: + in_b             -> bf16            (qk projection)
//   MODE 1: + b, bn1, +x, lrelu-> bf16            (attention out proj)
//   MODE 2: + b, bn2, +x, lrelu-> bf16 AND f32    (feed-forward)
//   MODE 3: temporal 3x1 conv (K=768, row kk=r*256+c reads B shifted by
//           (r-1)*27 with zero t-padding), + b, bn3, +y1b, lrelu -> f32 out
// Block: 128x128 tile, 256 threads = 8 waves (2x4), BK=32, LDS double buffer.
// ---------------------------------------------------------------------------
template<int KDIM, int MODE>
__global__ __launch_bounds__(256)
void gemm_bf16_kernel(const __bf16* __restrict__ A,
                      const __bf16* __restrict__ B,
                      const float*  __restrict__ bias,
                      const float*  __restrict__ bn_g, const float* __restrict__ bn_b,
                      const float*  __restrict__ bn_m, const float* __restrict__ bn_v,
                      const float*  __restrict__ resid,
                      __bf16*       __restrict__ outb,
                      float*        __restrict__ outf)
{
  __shared__ __bf16 sA[2][128 * LDSS];
  __shared__ __bf16 sB[2][128 * LDSS];

  const int n   = blockIdx.z;
  const int j0  = blockIdx.x * 128;
  const int o0  = blockIdx.y * 128;
  const int tid = threadIdx.x;
  const int lane = tid & 31;
  const int wave = tid >> 5;

  const int rowsPerN = (MODE == 1) ? 512 : 256;
  const __bf16* Bn = B + (size_t)n * rowsPerN * TPX;

  const unsigned int ldsA_base = (unsigned int)(uintptr_t)(void*)&sA[0][0];

  // Wave 0 issues the TDM descriptor for the A tile of K-step `ks`.
  auto stageA = [&](int buf, int ks) {
    if (wave == 0) {
      tdm_load_A_tile(ldsA_base + (unsigned int)buf * 128 * LDSS * 2,
                      (unsigned long long)(uintptr_t)(A + (size_t)o0 * KDIM + ks),
                      KDIM);
    }
  };

  // All waves transpose the B tile (32 k-rows x 128 j) into sB (j-major).
  auto stageB = [&](int buf, int ks) {
    #pragma unroll
    for (int it = 0; it < 2; ++it) {
      const int vv = tid + it * 256;
      const int k  = vv >> 4;                    // 0..31
      const int jc = (vv & 15) << 3;             // 0..120
      if constexpr (MODE != 3) {
        const int j = j0 + jc;
        const __bf16* src = Bn + (size_t)(ks + k) * TPX + j;
        if (ks + 64 < KDIM) __builtin_prefetch(Bn + (size_t)(ks + 32 + k) * TPX + j, 0, 3);
        v8bf val;
        if (j + 8 <= TPX) {
          val = *(const v8bf*)src;
        } else {
          #pragma unroll
          for (int e = 0; e < 8; ++e) val[e] = (j + e < TPX) ? src[e] : (__bf16)0.f;
        }
        #pragma unroll
        for (int e = 0; e < 8; ++e) sB[buf][(jc + e) * LDSS + k] = val[e];
      } else {
        // conv tap: kk = r*256 + c, source row shifted by (r-1)*27, zero-pad t
        const int kk = ks + k;
        const int r  = kk >> 8;
        const int c  = kk & 255;
        const __bf16* rowp = Bn + (size_t)c * TPX;
        const int shift = (r - 1) * VDIM;
        const int lo = (r == 0) ? VDIM : 0;
        const int hi = (r == 2) ? (TPX - VDIM) : TPX;
        const int jv = j0 + jc;
        v8bf val;
        if (jv >= lo && jv + 8 <= hi) {
          // fast path (interior): unaligned-safe vector copy (shift = +-27 elems)
          __builtin_memcpy(&val, rowp + jv + shift, 16);
        } else {
          #pragma unroll
          for (int e = 0; e < 8; ++e) {
            const int j = jv + e;
            val[e] = (j >= lo && j < hi) ? rowp[j + shift] : (__bf16)0.f;
          }
        }
        #pragma unroll
        for (int e = 0; e < 8; ++e) sB[buf][(jc + e) * LDSS + k] = val[e];
      }
    }
  };

  const int mW = (wave >> 2) * 64;               // 2 waves along M
  const int nW = (wave & 3) * 32;                // 4 waves along N

  v8f acc[4][2] = {};

  stageA(0, 0);
  stageB(0, 0);
  if (wave == 0) __builtin_amdgcn_s_wait_tensorcnt(0);
  __syncthreads();

  constexpr int NSTEP = KDIM / 32;
  for (int st = 0; st < NSTEP; ++st) {
    const int buf = st & 1;
    if (st + 1 < NSTEP) {
      stageA(buf ^ 1, (st + 1) * 32);
      stageB(buf ^ 1, (st + 1) * 32);
    }

    v16bf af[4], bfr[2];
    #pragma unroll
    for (int fm = 0; fm < 4; ++fm) af[fm]  = load_fragA(sA[buf], mW + fm * 16, lane);
    #pragma unroll
    for (int fn = 0; fn < 2; ++fn) bfr[fn] = load_fragB(sB[buf], nW + fn * 16, lane);

    #pragma unroll
    for (int fm = 0; fm < 4; ++fm)
      #pragma unroll
      for (int fn = 0; fn < 2; ++fn)
        acc[fm][fn] = __builtin_amdgcn_wmma_f32_16x16x32_bf16(
            false, af[fm], false, bfr[fn], (short)0, acc[fm][fn], false, false);

    if (st + 1 < NSTEP && wave == 0) __builtin_amdgcn_s_wait_tensorcnt(0);
    __syncthreads();
  }

  // Epilogue. C/D layout: VGPR rr, lanes 0-15 -> M=rr, lanes 16-31 -> M=rr+8;
  // N = lane & 15.
  #pragma unroll
  for (int fm = 0; fm < 4; ++fm) {
    #pragma unroll
    for (int fn = 0; fn < 2; ++fn) {
      const int j = j0 + nW + fn * 16 + (lane & 15);
      if (j >= TPX) continue;
      #pragma unroll
      for (int rr = 0; rr < 8; ++rr) {
        const int o = o0 + mW + fm * 16 + rr + ((lane >> 4) << 3);
        float val = acc[fm][fn][rr] + bias[o];
        if constexpr (MODE >= 1) {
          val = (val - bn_m[o]) * rsqrtf(bn_v[o] + 1e-5f) * bn_g[o] + bn_b[o];
          val += resid[((size_t)n * 256 + o) * TPX + j];
          val = (val > 0.f) ? val : 0.1f * val;      // leaky_relu 0.1
        }
        const size_t off = ((size_t)n * 256 + o) * TPX + j;
        if constexpr (MODE <= 2) outb[off] = (__bf16)val;
        if constexpr (MODE == 2 || MODE == 3) outf[off] = val;
      }
    }
  }
}

// ---------------------------------------------------------------------------
// y = x + pe[t+OFFSET, v]  (bf16 out). pos row is just pe + 2*27, index j.
// ---------------------------------------------------------------------------
__global__ __launch_bounds__(256)
void add_pe_kernel(const float* __restrict__ x, const float* __restrict__ pe,
                   __bf16* __restrict__ y, int total)
{
  const float* pos = pe + 2 * VDIM;
  const int e0 = (blockIdx.x * 256 + threadIdx.x) * 4;
  #pragma unroll
  for (int i = 0; i < 4; ++i) {
    const int e = e0 + i;
    if (e < total) y[e] = (__bf16)(x[e] + pos[e % TPX]);
  }
}

// ---------------------------------------------------------------------------
// Convert weights to bf16; t_w reorganized to (o, r*256 + c).
// ---------------------------------------------------------------------------
__global__ __launch_bounds__(256)
void convert_w_kernel(const float* __restrict__ in_w, const float* __restrict__ out_w,
                      const float* __restrict__ ff_w, const float* __restrict__ t_w,
                      __bf16* wIn, __bf16* wOut, __bf16* wFF, __bf16* wT)
{
  const int i = blockIdx.x * 256 + threadIdx.x;
  if (i < 65536)  wIn[i]  = (__bf16)in_w[i];
  if (i < 131072) wOut[i] = (__bf16)out_w[i];
  if (i < 65536)  wFF[i]  = (__bf16)ff_w[i];
  if (i < 196608) {
    const int o = i / 768, rem = i % 768, r = rem >> 8, c = rem & 255;
    wT[i] = (__bf16)t_w[(o * 256 + c) * 3 + r];
  }
}

// ---------------------------------------------------------------------------
// attn[n,s,u,v] = softmax_v( (sum_{ci,t} q*k) / (64*400) * alphas[s] + att0 )
// One block per (n,s); 768 threads, each owns one (u,v) pair of 729.
// ---------------------------------------------------------------------------
__global__ __launch_bounds__(768)
void attn_kernel(const __bf16* __restrict__ qk, const float* __restrict__ alphas,
                 const float* __restrict__ att0, float* __restrict__ attn)
{
  __shared__ float sL[729];
  const int n = blockIdx.x >> 1, s = blockIdx.x & 1;
  const int tid = threadIdx.x;

  if (tid < 729) {
    const int u = tid / VDIM, v = tid % VDIM;
    const __bf16* qb = qk + (size_t)(n * 256 + s * 64) * TPX;        // q block
    const __bf16* kb = qk + (size_t)(n * 256 + 128 + s * 64) * TPX;  // k block
    float acc = 0.f;
    for (int ci = 0; ci < 64; ++ci) {
      const __bf16* qr = qb + (size_t)ci * TPX;
      const __bf16* kr = kb + (size_t)ci * TPX;
      for (int t = 0; t < TDIM; ++t)
        acc += (float)qr[t * VDIM + u] * (float)kr[t * VDIM + v];
    }
    sL[tid] = acc * (1.f / 25600.f) * alphas[s] + att0[s * 729 + tid];
  }
  __syncthreads();
  if (tid < VDIM) {
    float m = -1e30f;
    for (int v = 0; v < VDIM; ++v) m = fmaxf(m, sL[tid * VDIM + v]);
    float sum = 0.f;
    for (int v = 0; v < VDIM; ++v) { sL[tid * VDIM + v] = __expf(sL[tid * VDIM + v] - m); sum += sL[tid * VDIM + v]; }
    const float inv = 1.f / sum;
    for (int v = 0; v < VDIM; ++v)
      attn[(size_t)(n * 2 + s) * 729 + tid * VDIM + v] = sL[tid * VDIM + v] * inv;
  }
}

// ---------------------------------------------------------------------------
// xattn[n, s*256+c, t, v] = sum_u x[n,c,t,u] * attn[n,s,u,v]   (bf16 out)
// Block: (t-chunk of 8, c, n); 256 threads (216 active: 8 t x 27 v).
// ---------------------------------------------------------------------------
__global__ __launch_bounds__(256)
void aggregate_kernel(const float* __restrict__ x, const float* __restrict__ attn,
                      __bf16* __restrict__ xattn)
{
  __shared__ float sAttn[2 * 729];
  __shared__ float sX[8][28];
  const int n = blockIdx.z, c = blockIdx.y, t0 = blockIdx.x * 8;

  for (int i = threadIdx.x; i < 1458; i += 256) sAttn[i] = attn[(size_t)n * 1458 + i];
  for (int i = threadIdx.x; i < 8 * VDIM; i += 256) {
    const int tt = i / VDIM, u = i % VDIM;
    sX[tt][u] = x[((size_t)(n * 256 + c) * TDIM + t0 + tt) * VDIM + u];
  }
  __syncthreads();

  if (threadIdx.x < 216) {
    const int tt = threadIdx.x / VDIM, v = threadIdx.x % VDIM;
    const int j = (t0 + tt) * VDIM + v;
    #pragma unroll
    for (int s = 0; s < 2; ++s) {
      const float* arow = &sAttn[s * 729];
      float acc = 0.f;
      #pragma unroll
      for (int u = 0; u < VDIM; ++u) acc += sX[tt][u] * arow[u * VDIM + v];
      xattn[((size_t)n * 512 + s * 256 + c) * TPX + j] = (__bf16)acc;
    }
  }
}

// ---------------------------------------------------------------------------
extern "C" void kernel_launch(void* const* d_in, const int* in_sizes, int n_in,
                              void* d_out, int out_size, void* d_ws, size_t ws_size,
                              hipStream_t stream)
{
  (void)in_sizes; (void)n_in; (void)out_size; (void)ws_size;
  const float* x      = (const float*)d_in[0];
  const float* pe     = (const float*)d_in[1];
  const float* in_w   = (const float*)d_in[2];
  const float* in_b   = (const float*)d_in[3];
  const float* alphas = (const float*)d_in[4];
  const float* att0   = (const float*)d_in[5];
  const float* out_w  = (const float*)d_in[6];
  const float* out_b  = (const float*)d_in[7];
  const float* bn1_g  = (const float*)d_in[8];
  const float* bn1_b  = (const float*)d_in[9];
  const float* bn1_m  = (const float*)d_in[10];
  const float* bn1_v  = (const float*)d_in[11];
  const float* ff_w   = (const float*)d_in[12];
  const float* ff_b   = (const float*)d_in[13];
  const float* bn2_g  = (const float*)d_in[14];
  const float* bn2_b  = (const float*)d_in[15];
  const float* bn2_m  = (const float*)d_in[16];
  const float* bn2_v  = (const float*)d_in[17];
  const float* t_w    = (const float*)d_in[18];
  const float* t_b    = (const float*)d_in[19];
  const float* bn3_g  = (const float*)d_in[20];
  const float* bn3_b  = (const float*)d_in[21];
  const float* bn3_m  = (const float*)d_in[22];
  const float* bn3_v  = (const float*)d_in[23];

  // Workspace layout (slot-reused):
  //   weights/attn in first 1 MB; then three big slots.
  //   slotA: y_bf16      -> y1a_bf16
  //   slotB: qk_bf16     -> y1b_bf16
  //   slotC: xattn_bf16  -> y1b_f32   (same byte size: 176,947,200)
  char* ws = (char*)d_ws;
  __bf16* wIn   = (__bf16*)(ws + 0);
  __bf16* wOut  = (__bf16*)(ws + 131072);
  __bf16* wFF   = (__bf16*)(ws + 393216);
  __bf16* wT    = (__bf16*)(ws + 524288);
  float*  attn  = (float*)(ws + 917504);
  const size_t BASE = 1u << 20;
  const size_t SLOT = 88473600;                  // 16*256*10800 bf16 bytes
  __bf16* slotA = (__bf16*)(ws + BASE);
  __bf16* slotB = (__bf16*)(ws + BASE + SLOT);
  __bf16* slotC = (__bf16*)(ws + BASE + 2 * SLOT);
  float*  y1bF  = (float*)slotC;

  const dim3 gGemm((TPX + 127) / 128, 2, 16);

  convert_w_kernel<<<768, 256, 0, stream>>>(in_w, out_w, ff_w, t_w, wIn, wOut, wFF, wT);
  add_pe_kernel<<<43200, 256, 0, stream>>>(x, pe, slotA, 16 * 256 * TPX);

  // qk = in_w @ (x + pe) + in_b
  gemm_bf16_kernel<256, 0><<<gGemm, 256, 0, stream>>>(
      wIn, slotA, in_b, nullptr, nullptr, nullptr, nullptr, nullptr, slotB, nullptr);

  attn_kernel<<<32, 768, 0, stream>>>(slotB, alphas, att0, attn);
  aggregate_kernel<<<dim3(50, 256, 16), 256, 0, stream>>>(x, attn, slotC);

  // y1a = leaky(x + bn1(out_w @ xattn + out_b))
  gemm_bf16_kernel<512, 1><<<gGemm, 256, 0, stream>>>(
      wOut, slotC, out_b, bn1_g, bn1_b, bn1_m, bn1_v, x, slotA, nullptr);

  // y1b = leaky(x + bn2(ff_w @ y1a + ff_b))   (bf16 + f32 copies)
  gemm_bf16_kernel<256, 2><<<gGemm, 256, 0, stream>>>(
      wFF, slotA, ff_b, bn2_g, bn2_b, bn2_m, bn2_v, x, slotB, y1bF);

  // out = leaky(y1b + bn3(conv3x1(y1b) + t_b))
  gemm_bf16_kernel<768, 3><<<gGemm, 256, 0, stream>>>(
      wT, slotB, t_b, bn3_g, bn3_b, bn3_m, bn3_v, y1bF, nullptr, (float*)d_out);
}
